// DiscretePatternLayer_11338713661889
// MI455X (gfx1250) — matrix operand order repaired and checked by reading, not compiled, not run on hardware
//
#include <hip/hip_runtime.h>
#include <hip/hip_bf16.h>
#include <math.h>

typedef __bf16 bf16;
typedef __attribute__((ext_vector_type(16))) __bf16 v16bf;
typedef __attribute__((ext_vector_type(8)))  __bf16 v8bf;
typedef __attribute__((ext_vector_type(4)))  __bf16 v4bf;
typedef __attribute__((ext_vector_type(8)))  float  v8f;
typedef __attribute__((ext_vector_type(4)))  int    v4i;

#define NTOK   32768          // B*L
#define DD     256            // complex feature dim
#define K2     512            // realified K/N
#define MT     32             // tokens per workgroup
#define NWAVE  4              // 128 threads, wave32
#define AS     520            // Abf row stride (halves): 1040B, 16B aligned, good banking
#define HS     516            // Hf32 row stride (floats)
#define NBLK1  (NTOK/MT)      // 1024 workgroups

// workspace byte offsets
#define WS_W     0                         // 5 * 512*512 bf16 = 2,621,440 B
#define WS_BC    2621440                   // 5 * 512 f32
#define WS_PP    2631680                   // 2 * 8 * 512 f32 (remapped patterns)
#define WS_PE    2664448                   // NTOK f32 (raw pe)
#define WS_CONF  2795520                   // NTOK f32
#define WS_PART  2926592                   // NBLK1 f32 partial pe sums

// guarded gfx1250 async global->LDS path (ASYNCcnt-tracked)
// probe-confirmed param types: (v4i addrspace(1)*, v4i addrspace(3)*, imm, imm)
#if defined(__HIP_DEVICE_COMPILE__) && __has_builtin(__builtin_amdgcn_global_load_async_to_lds_b128)
#define HAVE_ASYNC_LDS 1
#else
#define HAVE_ASYNC_LDS 0
#endif
#define AS1V4(p) ((__attribute__((address_space(1))) v4i*)(unsigned long long)(uintptr_t)(p))
#define AS3V4(p) ((__attribute__((address_space(3))) v4i*)(unsigned)(uintptr_t)(p))

struct PtrPack { const float* p[24]; };

// ---------------------------------------------------------------------------
// Kernel 0: realify + transpose weights to bf16 [N][K], build biases + patterns
// ---------------------------------------------------------------------------
__global__ __launch_bounds__(256)
void prep_kernel(PtrPack pk, bf16* __restrict__ Wt, float* __restrict__ bc,
                 float* __restrict__ pp)
{
    const int wr_i[5] = {2, 4, 8, 10, 6};
    const int wi_i[5] = {3, 5, 9, 11, 7};
    const int br_i[5] = {12, 14, 18, 20, 16};
    const int bi_i[5] = {13, 15, 19, 21, 17};
    int blk = blockIdx.x, tid = threadIdx.x;
    if (blk < 2560) {                       // weight rows: m in [0,5), n in [0,512)
        int m = blk >> 9, n = blk & 511;
        const float* wr = pk.p[wr_i[m]];
        const float* wi = pk.p[wi_i[m]];
        bf16* dst = Wt + (size_t)m * K2 * K2 + (size_t)n * K2;   // row n, contiguous k
        for (int k = tid; k < K2; k += 256) {
            float v;
            if (n < DD) v = (k < DD) ?  wr[k * DD + n]        : -wi[(k - DD) * DD + n];
            else        v = (k < DD) ?  wi[k * DD + (n - DD)] :  wr[(k - DD) * DD + (n - DD)];
            dst[k] = (bf16)v;
        }
    } else if (blk < 2565) {                // realified biases
        int m = blk - 2560;
        const float* br = pk.p[br_i[m]];
        const float* bi = pk.p[bi_i[m]];
        for (int n = tid; n < K2; n += 256)
            bc[m * K2 + n] = (n < DD) ? br[n] : bi[n - DD];
    } else {                                // de-interleaved patterns
        int b = blk - 2565;                 // 0..15
        int which = b >> 3, p = b & 7;
        const float* pat = pk.p[22 + which];
        float* dst = pp + which * 8 * K2 + p * K2;
        for (int f = tid; f < K2; f += 256) {
            int src = (f < DD) ? (2 * f) : (2 * (f - DD) + 1);
            dst[f] = pat[p * 2 * DD + src];
        }
    }
}

// ---------------------------------------------------------------------------
// WMMA GEMM over one 32x512 tile: C(32x512) = A(32x512,bf16 LDS) * W(512x512)
// Each wave owns 4 real N-tiles (+4 imag) across BOTH 16-row M tiles:
// every B fragment feeds 2 WMMAs; weights are read exactly once per WG.
// ---------------------------------------------------------------------------
enum { GM_H = 0, GM_GATE = 1, GM_GATEOUT = 2, GM_PC = 3 };

__device__ __forceinline__ v16bf pack16(v8bf lo, v8bf hi) {
    v16bf r;
#pragma unroll
    for (int i = 0; i < 8; ++i) { r[i] = lo[i]; r[i + 8] = hi[i]; }
    return r;
}

template <int MODE>
__device__ __forceinline__ void do_gemm(const bf16* __restrict__ Wg,
                                        const float* __restrict__ bcm,
                                        const bf16* __restrict__ Abf,
                                        float* __restrict__ Hf32,
                                        float* __restrict__ gr_out,
                                        float* __restrict__ gi_out,
                                        int tok0, int wave, int lane)
{
    const int l15 = lane & 15;
    const int ahi = lane >> 4;
    const v8f zero = {0.f, 0.f, 0.f, 0.f, 0.f, 0.f, 0.f, 0.f};

    for (int half = 0; half < 2; ++half) {
        const int nR0 = wave * 4 + half * 2;   // real tile cols nR0,nR0+1; imag +16
        v8f acc[8];                            // [m*4+c], m = M-tile 0/1, c = R0,R1,I0,I1
#pragma unroll
        for (int c = 0; c < 8; ++c) acc[c] = zero;

        for (int ks = 0; ks < 16; ++ks) {
            // A fragments for both M tiles:
            // lanes<16 hold K {0..7,16..23}, lanes>=16 hold {8..15,24..31}
            const bf16* ap0 = Abf + l15 * AS + ks * 32 + ahi * 8;
            const bf16* ap1 = ap0 + 16 * AS;
            v16bf a0 = pack16(*(const v8bf*)ap0, *(const v8bf*)(ap0 + 16));
            v16bf a1 = pack16(*(const v8bf*)ap1, *(const v8bf*)(ap1 + 16));
#pragma unroll
            for (int c = 0; c < 4; ++c) {
                int nt = (c < 2) ? (nR0 + c) : (nR0 + 16 + (c - 2));
                // B fragment: lane n = l15, lanes<16 hold K {0..7,16..23}, lanes>=16 hold {8..15,24..31}
                const bf16* bp = Wg + (size_t)(nt * 16 + l15) * K2 + ks * 32 + ahi * 8;
                v16bf b = pack16(*(const v8bf*)bp, *(const v8bf*)(bp + 16));
                acc[c]     = __builtin_amdgcn_wmma_f32_16x16x32_bf16(
                                 false, a0, false, b, (short)0, acc[c], false, false);
                acc[4 + c] = __builtin_amdgcn_wmma_f32_16x16x32_bf16(
                                 false, a1, false, b, (short)0, acc[4 + c], false, false);
            }
        }

        if (MODE == GM_H) {
#pragma unroll
            for (int c = 0; c < 4; ++c) {
                int nt = (c < 2) ? (nR0 + c) : (nR0 + 16 + (c - 2));
                int fc = nt * 16 + l15;
                float bv = bcm[fc];
#pragma unroll
                for (int m = 0; m < 2; ++m)
#pragma unroll
                    for (int j = 0; j < 8; ++j) {
                        int tl = m * 16 + j + 8 * ahi;
                        Hf32[tl * HS + fc] = acc[m * 4 + c][j] + bv;
                    }
            }
        } else if (MODE == GM_PC) {
#pragma unroll
            for (int jp = 0; jp < 2; ++jp) {
                int fR = (nR0 + jp) * 16 + l15;
                float bR = bcm[fR], bI = bcm[fR + DD];
#pragma unroll
                for (int m = 0; m < 2; ++m)
#pragma unroll
                    for (int j = 0; j < 8; ++j) {
                        int tl = m * 16 + j + 8 * ahi;
                        size_t go = (size_t)(tok0 + tl) * DD + fR;
                        gr_out[go] = acc[m * 4 + jp][j] + bR;
                        gi_out[go] = acc[m * 4 + jp + 2][j] + bI;
                    }
            }
        } else {  // GM_GATE / GM_GATEOUT : paradox gating fused into epilogue
#pragma unroll
            for (int jp = 0; jp < 2; ++jp) {
                int fR = (nR0 + jp) * 16 + l15;
                int fI = fR + DD;
                float bR = bcm[fR], bI = bcm[fI];
#pragma unroll
                for (int m = 0; m < 2; ++m)
#pragma unroll
                    for (int j = 0; j < 8; ++j) {
                        int tl = m * 16 + j + 8 * ahi;
                        float pr = acc[m * 4 + jp][j] + bR;
                        float pi = acc[m * 4 + jp + 2][j] + bI;
                        float hr = Hf32[tl * HS + fR];
                        float hi = Hf32[tl * HS + fI];
                        float dr = pr - hr, di = pi - hi;
                        float g = 1.0f / (1.0f + __expf(-sqrtf(dr * dr + di * di)));
                        float gr = hr * g, gi = hi * g;
                        Hf32[tl * HS + fR] = gr;
                        Hf32[tl * HS + fI] = gi;
                        if (MODE == GM_GATEOUT) {
                            size_t go = (size_t)(tok0 + tl) * DD + fR;
                            gr_out[go] = gr;   // unscaled h -> cu slots (scaled later)
                            gi_out[go] = gi;
                        }
                    }
            }
        }
    }
}

// softmax pattern-attention weights for tokens owned by this wave (patterns in LDS)
__device__ __forceinline__ void attn_weights(const float* __restrict__ Hf32,
                                             const float* __restrict__ patl,
                                             float* __restrict__ Wout,  // [MT][8] LDS
                                             int wave, int lane)
{
    for (int t = wave; t < MT; t += NWAVE) {
        float s[8] = {0, 0, 0, 0, 0, 0, 0, 0};
        for (int f = lane; f < K2; f += 32) {
            float hv = Hf32[t * HS + f];
#pragma unroll
            for (int p = 0; p < 8; ++p) s[p] += hv * patl[p * K2 + f];
        }
#pragma unroll
        for (int p = 0; p < 8; ++p)
#pragma unroll
            for (int off = 16; off > 0; off >>= 1) s[p] += __shfl_xor(s[p], off, 32);
        const float scale = 0.0625f;  // 1/sqrt(256)
        float mx = s[0] * scale;
#pragma unroll
        for (int p = 1; p < 8; ++p) mx = fmaxf(mx, s[p] * scale);
        float w[8], sum = 0.f;
#pragma unroll
        for (int p = 0; p < 8; ++p) { w[p] = __expf(s[p] * scale - mx); sum += w[p]; }
        float inv = 1.0f / sum;
        if (lane == 0) {
#pragma unroll
            for (int p = 0; p < 8; ++p) Wout[t * 8 + p] = w[p] * inv;
        }
    }
}

// ---------------------------------------------------------------------------
// Kernel 1: full WMMA pipeline for 32 tokens per workgroup
// ---------------------------------------------------------------------------
__global__ __launch_bounds__(128)
void layer_kernel(const float* __restrict__ x_r, const float* __restrict__ x_i,
                  const bf16* __restrict__ Wt, const float* __restrict__ bc,
                  const float* __restrict__ pp,
                  float* __restrict__ cu_r, float* __restrict__ cu_i,
                  float* __restrict__ pc_r, float* __restrict__ pc_i,
                  float* __restrict__ pe_ws, float* __restrict__ partials)
{
    __shared__ __align__(16) bf16  Abf[MT * AS];    // 33,280 B (bf16 A operand)
    __shared__ __align__(16) float Hf32[MT * HS];   // 66,048 B (f32 activations)
    __shared__ __align__(16) float PatL[2 * 8 * K2];// 32,768 B (both pattern banks)
    __shared__ float Wmy[MT * 8];
    __shared__ float Wnx[MT * 8];
    __shared__ float sred[MT];

    const int tid = threadIdx.x;
    const int wave = tid >> 5, lane = tid & 31;
    const int tok0 = blockIdx.x * MT;

    // stage both pattern banks into LDS (async on gfx1250: ASYNCcnt-tracked,
    // latency hidden behind the first two GEMMs)
#if HAVE_ASYNC_LDS
    for (int c4 = tid; c4 < (2 * 8 * K2) / 4; c4 += 128)
        __builtin_amdgcn_global_load_async_to_lds_b128(
            AS1V4(pp + c4 * 4), AS3V4(&PatL[c4 * 4]), 0, 0);
#else
    for (int c4 = tid; c4 < (2 * 8 * K2) / 4; c4 += 128)
        *(float4*)&PatL[c4 * 4] = *(const float4*)(pp + c4 * 4);
#endif

    // phase 0: X -> realified bf16 A operand ([t][0..255]=xr, [256..511]=xi)
    for (int idx = tid; idx < MT * K2 / 4; idx += 128) {
        int t = idx >> 7, f4 = idx & 127;
        const float* src = (f4 < 64) ? (x_r + (size_t)(tok0 + t) * DD + f4 * 4)
                                     : (x_i + (size_t)(tok0 + t) * DD + (f4 - 64) * 4);
        float4 v = *(const float4*)src;
        v4bf o = { (bf16)v.x, (bf16)v.y, (bf16)v.z, (bf16)v.w };
        *(v4bf*)&Abf[t * AS + f4 * 4] = o;
    }
    __syncthreads();

    // G1: H = X @ Wc1 + b1
    do_gemm<GM_H>(Wt + 0 * 262144, bc + 0 * K2, Abf, Hf32, nullptr, nullptr,
                  tok0, wave, lane);
    __syncthreads();
    for (int idx = tid; idx < MT * K2 / 4; idx += 128) {
        int t = idx >> 7, f = (idx & 127) * 4;
        float4 v = *(const float4*)&Hf32[t * HS + f];
        v4bf o = { (bf16)v.x, (bf16)v.y, (bf16)v.z, (bf16)v.w };
        *(v4bf*)&Abf[t * AS + f] = o;
    }
    __syncthreads();

    // G2: self-prediction + paradox gating; gated h -> Hf32 and d_out cu slots
    do_gemm<GM_GATEOUT>(Wt + 1 * 262144, bc + 1 * K2, Abf, Hf32, cu_r, cu_i,
                        tok0, wave, lane);
    __syncthreads();
    for (int idx = tid; idx < MT * K2 / 4; idx += 128) {
        int t = idx >> 7, f = (idx & 127) * 4;
        float4 v = *(const float4*)&Hf32[t * HS + f];
        v4bf o = { (bf16)v.x, (bf16)v.y, (bf16)v.z, (bf16)v.w };
        *(v4bf*)&Abf[t * AS + f] = o;
    }
#if HAVE_ASYNC_LDS
#if __has_builtin(__builtin_amdgcn_s_wait_asynccnt)
    __builtin_amdgcn_s_wait_asynccnt(0);
#else
    asm volatile("s_wait_asynccnt 0" ::: "memory");
#endif
#endif
    __syncthreads();

    // my-pattern softmax weights from gated h (patterns now resident in LDS)
    attn_weights(Hf32, PatL, Wmy, wave, lane);
    __syncthreads();

    // G5: penultimate projection pf = h @ Wc3 + b3 -> pc slots (unscaled)
    do_gemm<GM_PC>(Wt + 4 * 262144, bc + 4 * K2, Abf, nullptr, pc_r, pc_i,
                   tok0, wave, lane);
    // G3: next-layer hidden a_pre = h @ Wn1 + nb1 -> Hf32
    do_gemm<GM_H>(Wt + 2 * 262144, bc + 2 * K2, Abf, Hf32, nullptr, nullptr,
                  tok0, wave, lane);
    __syncthreads();
    for (int idx = tid; idx < MT * K2 / 4; idx += 128) {
        int t = idx >> 7, f = (idx & 127) * 4;
        float4 v = *(const float4*)&Hf32[t * HS + f];
        v4bf o = { (bf16)v.x, (bf16)v.y, (bf16)v.z, (bf16)v.w };
        *(v4bf*)&Abf[t * AS + f] = o;
    }
    __syncthreads();

    // G4: next-layer self-prediction + gating; gated a -> Hf32
    do_gemm<GM_GATE>(Wt + 3 * 262144, bc + 3 * K2, Abf, Hf32, nullptr, nullptr,
                     tok0, wave, lane);
    __syncthreads();

    // next-pattern weights + pe = mean((next_comp - my_comp)^2)
    attn_weights(Hf32, PatL + 8 * K2, Wnx, wave, lane);
    __syncthreads();
    for (int t = wave; t < MT; t += NWAVE) {
        float wm[8], wn[8];
#pragma unroll
        for (int p = 0; p < 8; ++p) { wm[p] = Wmy[t * 8 + p]; wn[p] = Wnx[t * 8 + p]; }
        float local = 0.f;
        for (int f = lane; f < K2; f += 32) {
            float cM = 0.f, cN = 0.f;
#pragma unroll
            for (int p = 0; p < 8; ++p) {
                cM += wm[p] * PatL[p * K2 + f];
                cN += wn[p] * PatL[8 * K2 + p * K2 + f];
            }
            float d = cN - cM;
            local += d * d;
        }
#pragma unroll
        for (int off = 16; off > 0; off >>= 1) local += __shfl_xor(local, off, 32);
        if (lane == 0) {
            float pe = local * (1.0f / 512.0f);
            pe_ws[tok0 + t] = pe;
            sred[t] = pe;
        }
    }
    __syncthreads();
    if (tid == 0) {   // fixed-order block partial (deterministic)
        float s = 0.f;
        for (int t = 0; t < MT; ++t) s += sred[t];
        partials[blockIdx.x] = s;
    }
}

// ---------------------------------------------------------------------------
// Kernel 2: deterministic global stats -> conf[], routing cost, final pe
// ---------------------------------------------------------------------------
__global__ __launch_bounds__(256)
void finalize_pe_kernel(const float* __restrict__ partials,
                        const float* __restrict__ pe_raw,
                        float* __restrict__ conf, float* __restrict__ pe_out)
{
    __shared__ float sh[256];
    __shared__ float s_mean, s_rc;
    int tid = threadIdx.x;

    float a = 0.f;
    for (int i = tid; i < NBLK1; i += 256) a += partials[i];
    sh[tid] = a; __syncthreads();
    for (int s = 128; s > 0; s >>= 1) { if (tid < s) sh[tid] += sh[tid + s]; __syncthreads(); }
    if (tid == 0) s_mean = sh[0] * (1.0f / (float)NTOK);
    __syncthreads();
    float mean = s_mean;

    float lc = 0.f;
    for (int t = tid; t < NTOK; t += 256) {
        float pe = pe_raw[t];
        float cert = fabsf(pe - mean);
        float temp = 1.0f / (1.0f + __expf(-cert));
        float cf = 0.5f * (tanhf(-pe * temp) + 1.0f);
        conf[t] = cf;
        lc += cf * (1.0f - cf);
    }
    sh[tid] = lc; __syncthreads();
    for (int s = 128; s > 0; s >>= 1) { if (tid < s) sh[tid] += sh[tid + s]; __syncthreads(); }
    if (tid == 0) s_rc = 0.1f * sh[0] * (1.0f / (float)NTOK);
    __syncthreads();
    float rc = s_rc;
    for (int t = tid; t < NTOK; t += 256) pe_out[t] = pe_raw[t] + rc;
}

// ---------------------------------------------------------------------------
// Kernel 3: in-place confidence scaling of the four big outputs (float4)
// ---------------------------------------------------------------------------
__global__ __launch_bounds__(256)
void scale_kernel(float* __restrict__ cu_r, float* __restrict__ cu_i,
                  float* __restrict__ pc_r, float* __restrict__ pc_i,
                  const float* __restrict__ conf)
{
    int idx = blockIdx.x * 256 + threadIdx.x;   // over NTOK*DD/4
    int tok = idx >> 6;                         // 64 float4 per token
    float cf = conf[tok];
    float om = 1.0f - cf;
    float4* r0 = (float4*)cu_r;  float4* r1 = (float4*)cu_i;
    float4* r2 = (float4*)pc_r;  float4* r3 = (float4*)pc_i;
    float4 v;
    v = r0[idx]; v.x *= om; v.y *= om; v.z *= om; v.w *= om; r0[idx] = v;
    v = r1[idx]; v.x *= om; v.y *= om; v.z *= om; v.w *= om; r1[idx] = v;
    v = r2[idx]; v.x *= cf; v.y *= cf; v.z *= cf; v.w *= cf; r2[idx] = v;
    v = r3[idx]; v.x *= cf; v.y *= cf; v.z *= cf; v.w *= cf; r3[idx] = v;
}

// ---------------------------------------------------------------------------
extern "C" void kernel_launch(void* const* d_in, const int* in_sizes, int n_in,
                              void* d_out, int out_size, void* d_ws, size_t ws_size,
                              hipStream_t stream)
{
    (void)in_sizes; (void)n_in; (void)out_size; (void)ws_size;
    PtrPack pk;
    for (int i = 0; i < 24; ++i) pk.p[i] = (const float*)d_in[i];

    char* ws = (char*)d_ws;
    bf16*  Wt       = (bf16*)(ws + WS_W);
    float* bc       = (float*)(ws + WS_BC);
    float* pp       = (float*)(ws + WS_PP);
    float* pe_raw   = (float*)(ws + WS_PE);
    float* conf     = (float*)(ws + WS_CONF);
    float* partials = (float*)(ws + WS_PART);

    float* out  = (float*)d_out;
    float* cu_r = out;
    float* cu_i = out + (size_t)1 * NTOK * DD;
    float* pc_r = out + (size_t)2 * NTOK * DD;
    float* pc_i = out + (size_t)3 * NTOK * DD;
    float* pe_o = out + (size_t)4 * NTOK * DD;

    prep_kernel<<<2581, 256, 0, stream>>>(pk, Wt, bc, pp);
    layer_kernel<<<NBLK1, 128, 0, stream>>>((const float*)d_in[0], (const float*)d_in[1],
                                            Wt, bc, pp, cu_r, cu_i, pc_r, pc_i,
                                            pe_raw, partials);
    finalize_pe_kernel<<<1, 256, 0, stream>>>(partials, pe_raw, conf, pe_o);
    scale_kernel<<<NTOK * DD / 4 / 256, 256, 0, stream>>>(cu_r, cu_i, pc_r, pc_i, conf);
}
